// PlanningAttention_49452253446293
// MI455X (gfx1250) — 8-bit matrix operand order repaired and checked by reading, compiled, not run on hardware
//
#include <hip/hip_runtime.h>
#include <cstdint>
#include <cstddef>

#define BB 8
#define SS 8192
#define QD 1024
#define KD 1024
#define AD 1024
#define CD 2048
#define EPSF 1e-5f
#define NCHUNK 64
#define TOKC 128

typedef int v8i __attribute__((ext_vector_type(8)));
union V8U { v8i v; int e[8]; };

static constexpr size_t WS_PART   = 0;
static constexpr size_t WS_SCALES = 4096;
static constexpr size_t WS_GQ     = 4096 + 64;
static constexpr size_t WS_G2     = 4096 + 128;
static constexpr size_t WS_OFFS   = 4096 + 192;
static constexpr size_t WS_AQQ    = 8192;
static constexpr size_t WS_AQC    = 8192 + 16384;
static constexpr size_t WS_WQQ    = 65536;
static constexpr size_t WS_WQK    = WS_WQQ + ((size_t)1 << 20);
static constexpr size_t WS_WQVT   = WS_WQK + ((size_t)1 << 20);
static constexpr size_t WS_WQC    = WS_WQVT + ((size_t)1 << 20);
static constexpr size_t WS_QPROJ  = WS_WQC + ((size_t)2 << 20);
static constexpr size_t WS_P      = WS_QPROJ + 32768;
static constexpr size_t WS_UNORM  = WS_P + 32768;
static constexpr size_t WS_CTX    = WS_UNORM + 32768;
static constexpr size_t WS_PARTM  = WS_CTX + 32768;
static constexpr size_t WS_PARTL  = WS_PARTM + 4096;
static constexpr size_t WS_PARTU  = WS_PARTL + 4096;

__device__ __forceinline__ float wred_max(float v) {
  #pragma unroll
  for (int o = 16; o > 0; o >>= 1) v = fmaxf(v, __shfl_xor(v, o, 32));
  return v;
}
__device__ __forceinline__ float wred_sum(float v) {
  #pragma unroll
  for (int o = 16; o > 0; o >>= 1) v += __shfl_xor(v, o, 32);
  return v;
}

__global__ void k_abs_partial(const float* __restrict__ wq, const float* __restrict__ wk,
                              const float* __restrict__ wv, const float* __restrict__ wc,
                              float* __restrict__ part) {
  __shared__ float red[256];
  const int widx = blockIdx.y;
  const float* w = (widx == 0) ? wq : (widx == 1) ? wk : (widx == 2) ? wv : wc;
  const int n = (widx == 3) ? (QD * CD) : (QD * KD);
  const int tid = threadIdx.x;
  float s = 0.f;
  for (int i = blockIdx.x * 256 + tid; i < n; i += gridDim.x * 256) s += fabsf(w[i]);
  red[tid] = s; __syncthreads();
  for (int st = 128; st > 0; st >>= 1) { if (tid < st) red[tid] += red[tid + st]; __syncthreads(); }
  if (tid == 0) part[widx * 256 + blockIdx.x] = red[0];
}

__global__ void k_scales(const float* __restrict__ part, float* __restrict__ scales) {
  __shared__ float red[256];
  const int widx = blockIdx.x, tid = threadIdx.x;
  const float n = (widx == 3) ? (float)(QD * CD) : (float)(QD * KD);
  red[tid] = part[widx * 256 + tid]; __syncthreads();
  for (int st = 128; st > 0; st >>= 1) { if (tid < st) red[tid] += red[tid + st]; __syncthreads(); }
  if (tid == 0) scales[widx] = red[0] / n + EPSF;
}

__global__ void k_quant_w(const float* __restrict__ wq, const float* __restrict__ wk,
                          const float* __restrict__ wv, const float* __restrict__ wc,
                          const float* __restrict__ scales,
                          signed char* __restrict__ oq, signed char* __restrict__ ok,
                          signed char* __restrict__ ovT, signed char* __restrict__ oc) {
  const int widx = blockIdx.y;
  const float* w = (widx == 0) ? wq : (widx == 1) ? wk : (widx == 2) ? wv : wc;
  const int n = (widx == 3) ? (QD * CD) : (QD * KD);
  const float inv = 1.f / scales[widx];
  for (int i = blockIdx.x * 256 + threadIdx.x; i < n; i += gridDim.x * 256) {
    float t = fminf(fmaxf(rintf(w[i] * inv), -1.f), 1.f);
    signed char c = (signed char)t;
    if (widx == 0) oq[i] = c;
    else if (widx == 1) ok[i] = c;
    else if (widx == 2) { int o = i >> 10, d = i & 1023; ovT[(size_t)d * QD + o] = c; }
    else oc[i] = c;
  }
}

__global__ void k_quant_query(const float* __restrict__ q, signed char* __restrict__ aq,
                              float* __restrict__ gq) {
  __shared__ float red[256]; __shared__ float gsh;
  const int b = blockIdx.x, tid = threadIdx.x;
  float amax = 0.f;
  for (int i = tid; i < QD; i += 256) amax = fmaxf(amax, fabsf(q[b * QD + i]));
  red[tid] = amax; __syncthreads();
  for (int st = 128; st > 0; st >>= 1) { if (tid < st) red[tid] = fmaxf(red[tid], red[tid + st]); __syncthreads(); }
  if (tid == 0) { gsh = red[0] + EPSF; gq[b] = gsh; }
  __syncthreads();
  const float qs = 127.f / gsh;
  for (int i = tid; i < QD; i += 256) {
    float v = fminf(fmaxf(rintf(q[b * QD + i] * qs), -128.f), 127.f);
    aq[b * QD + i] = (signed char)v;
    aq[(b + 8) * QD + i] = 0;
  }
}

__global__ void k_wmma_proj(const signed char* __restrict__ A, const signed char* __restrict__ Bw,
                            const float* __restrict__ gvec, const float* __restrict__ scales,
                            int sidx, const float* __restrict__ bias,
                            float* __restrict__ out, int K) {
  const int lane = threadIdx.x;
  const int h = lane >> 4, mn = lane & 15;
  const int col = blockIdx.x * 16 + mn;
  const float s = scales[sidx];
  V8U acc;
  #pragma unroll
  for (int r = 0; r < 8; ++r) acc.e[r] = 0;
  for (int k0 = 0; k0 < K; k0 += 64) {
    V8U a, bf;
    #pragma unroll
    for (int v = 0; v < 8; ++v) {
      const int kA = k0 + ((v >> 2) * 32) + (((v >> 1) & 1) * 16) + ((v & 1) * 4) + h * 8;
      const int kB = k0 + ((v >> 2) * 32) + (((v >> 1) & 1) * 16) + ((v & 1) * 4) + h * 8;
      a.e[v]  = *(const int*)(A  + (size_t)mn  * K + kA);
      bf.e[v] = *(const int*)(Bw + (size_t)col * K + kB);
    }
    acc.v = __builtin_amdgcn_wmma_i32_16x16x64_iu8(true, a.v, true, bf.v, acc.v, false, false);
  }
  if (h == 0) {
    #pragma unroll
    for (int r = 0; r < 8; ++r)
      out[(size_t)r * QD + col] = (float)acc.e[r] * (gvec[r] * (1.f / 127.f)) * s + bias[col];
  }
}

__global__ void k_offs(const float* __restrict__ qproj, const float* __restrict__ bk,
                       float* __restrict__ offs) {
  __shared__ float red[256];
  const int tid = threadIdx.x;
  for (int b = 0; b < BB; ++b) {
    float partial = 0.f;
    for (int a = tid; a < AD; a += 256) partial += qproj[b * AD + a] * bk[a];
    red[tid] = partial; __syncthreads();
    for (int st = 128; st > 0; st >>= 1) { if (tid < st) red[tid] += red[tid + st]; __syncthreads(); }
    if (tid == 0) offs[b] = red[0];
    __syncthreads();
  }
}

__global__ void k_pvec(const float* __restrict__ qproj, const signed char* __restrict__ wqk,
                       const float* __restrict__ scales, float* __restrict__ p) {
  __shared__ float qs[BB * AD];
  const int tid = threadIdx.x;
  for (int i = tid; i < BB * AD; i += 256) qs[i] = qproj[i];
  __syncthreads();
  const int d = blockIdx.x * 256 + tid;
  const float sk = scales[1];
  float acc[BB];
  #pragma unroll
  for (int b = 0; b < BB; ++b) acc[b] = 0.f;
  for (int a = 0; a < AD; ++a) {
    float wv = (float)wqk[(size_t)a * KD + d];
    #pragma unroll
    for (int b = 0; b < BB; ++b) acc[b] += qs[b * AD + a] * wv;
  }
  #pragma unroll
  for (int b = 0; b < BB; ++b) p[b * KD + d] = sk * acc[b];
}

__global__ void k_scan(const float* __restrict__ cmem, const float* __restrict__ p,
                       const float* __restrict__ offs,
                       float* __restrict__ part_m, float* __restrict__ part_l,
                       float* __restrict__ part_u) {
  __shared__ float p_sh[KD];
  __shared__ float u_sh[KD];
  __shared__ float wm[8], wl[8];
  const int tid = threadIdx.x, lane = tid & 31, w = tid >> 5;
  const int b = blockIdx.y, chunk = blockIdx.x;
  for (int i = tid; i < KD; i += 256) { p_sh[i] = p[b * KD + i]; u_sh[i] = 0.f; }
  __syncthreads();
  const float off_b = offs[b];
  float m = -3.0e38f, l = 0.f;
  float u[32];
  #pragma unroll
  for (int i = 0; i < 32; ++i) u[i] = 0.f;
  const size_t base = ((size_t)b * SS + (size_t)chunk * TOKC + (size_t)w * 16) * KD;
  for (int t = 0; t < 16; ++t) {
    const float* row = cmem + base + (size_t)t * KD;
    if (t < 15) {
      const char* nx = (const char*)(row + KD);
      __builtin_prefetch(nx + lane * 128, 0, 3);
    }
    float4 X[8];
    #pragma unroll
    for (int j = 0; j < 8; ++j) X[j] = ((const float4*)row)[lane + j * 32];
    float amax = 0.f;
    #pragma unroll
    for (int j = 0; j < 8; ++j) {
      amax = fmaxf(amax, fabsf(X[j].x)); amax = fmaxf(amax, fabsf(X[j].y));
      amax = fmaxf(amax, fabsf(X[j].z)); amax = fmaxf(amax, fabsf(X[j].w));
    }
    const float g = wred_max(amax) + EPSF;
    const float qsc = 127.f / g, dq = g * (1.f / 127.f);
    float sdot = 0.f;
    #pragma unroll
    for (int j = 0; j < 8; ++j) {
      const int d0 = lane * 4 + j * 128;
      X[j].x = fminf(fmaxf(rintf(X[j].x * qsc), -128.f), 127.f);
      X[j].y = fminf(fmaxf(rintf(X[j].y * qsc), -128.f), 127.f);
      X[j].z = fminf(fmaxf(rintf(X[j].z * qsc), -128.f), 127.f);
      X[j].w = fminf(fmaxf(rintf(X[j].w * qsc), -128.f), 127.f);
      sdot += p_sh[d0 + 0] * X[j].x + p_sh[d0 + 1] * X[j].y
            + p_sh[d0 + 2] * X[j].z + p_sh[d0 + 3] * X[j].w;
    }
    sdot = wred_sum(sdot);
    const float score = (sdot * dq + off_b) * 0.03125f;
    const float mn  = fmaxf(m, score);
    const float esc = __expf(m - mn);
    const float ew  = __expf(score - mn);
    l = l * esc + ew;
    const float f = ew * dq;
    #pragma unroll
    for (int j = 0; j < 8; ++j) {
      u[j * 4 + 0] = u[j * 4 + 0] * esc + f * X[j].x;
      u[j * 4 + 1] = u[j * 4 + 1] * esc + f * X[j].y;
      u[j * 4 + 2] = u[j * 4 + 2] * esc + f * X[j].z;
      u[j * 4 + 3] = u[j * 4 + 3] * esc + f * X[j].w;
    }
    m = mn;
  }
  if (lane == 0) { wm[w] = m; wl[w] = l; }
  __syncthreads();
  float M = wm[0];
  #pragma unroll
  for (int j = 1; j < 8; ++j) M = fmaxf(M, wm[j]);
  const float fac = __expf(m - M);
  #pragma unroll
  for (int j = 0; j < 8; ++j) {
    const int d0 = lane * 4 + j * 128;
    atomicAdd(&u_sh[d0 + 0], u[j * 4 + 0] * fac);
    atomicAdd(&u_sh[d0 + 1], u[j * 4 + 1] * fac);
    atomicAdd(&u_sh[d0 + 2], u[j * 4 + 2] * fac);
    atomicAdd(&u_sh[d0 + 3], u[j * 4 + 3] * fac);
  }
  __syncthreads();
  const int ci = b * NCHUNK + chunk;
  if (tid == 0) {
    float L = 0.f;
    for (int j = 0; j < 8; ++j) L += wl[j] * __expf(wm[j] - M);
    part_m[ci] = M; part_l[ci] = L;
  }
  for (int i = tid; i < KD; i += 256) part_u[(size_t)ci * KD + i] = u_sh[i];
}

__global__ void k_combine(const float* __restrict__ pm, const float* __restrict__ pl,
                          const float* __restrict__ pu, float* __restrict__ unorm) {
  __shared__ float facs[NCHUNK];
  __shared__ float Lsh;
  const int b = blockIdx.x, tid = threadIdx.x;
  float M = -3.0e38f;
  for (int j = 0; j < NCHUNK; ++j) M = fmaxf(M, pm[b * NCHUNK + j]);
  if (tid < NCHUNK) facs[tid] = __expf(pm[b * NCHUNK + tid] - M);
  __syncthreads();
  if (tid == 0) {
    float L = 0.f;
    for (int j = 0; j < NCHUNK; ++j) L += pl[b * NCHUNK + j] * facs[j];
    Lsh = L;
  }
  __syncthreads();
  const float invL = 1.f / Lsh;
  for (int d = tid; d < KD; d += 256) {
    float acc = 0.f;
    for (int j = 0; j < NCHUNK; ++j) acc += pu[((size_t)(b * NCHUNK + j)) * KD + d] * facs[j];
    unorm[b * KD + d] = acc * invL;
  }
}

__global__ void k_ctx(const float* __restrict__ unorm, const signed char* __restrict__ wqvT,
                      const float* __restrict__ scales, const float* __restrict__ bv,
                      float* __restrict__ ctxv) {
  __shared__ float us[BB * KD];
  const int tid = threadIdx.x;
  for (int i = tid; i < BB * KD; i += 256) us[i] = unorm[i];
  __syncthreads();
  const int o = blockIdx.x * 256 + tid;
  const float sv = scales[2];
  float acc[BB];
  #pragma unroll
  for (int b = 0; b < BB; ++b) acc[b] = 0.f;
  for (int d = 0; d < KD; ++d) {
    float wv = (float)wqvT[(size_t)d * QD + o];
    #pragma unroll
    for (int b = 0; b < BB; ++b) acc[b] += us[b * KD + d] * wv;
  }
  #pragma unroll
  for (int b = 0; b < BB; ++b) ctxv[b * QD + o] = sv * acc[b] + bv[o];
}

__global__ void k_quant_combined(const float* __restrict__ query, const float* __restrict__ ctxv,
                                 signed char* __restrict__ aqc, float* __restrict__ g2) {
  __shared__ float red[256]; __shared__ float gsh;
  const int b = blockIdx.x, tid = threadIdx.x;
  float amax = 0.f;
  for (int i = tid; i < CD; i += 256) {
    float v = (i < QD) ? query[b * QD + i] : ctxv[b * QD + (i - QD)];
    amax = fmaxf(amax, fabsf(v));
  }
  red[tid] = amax; __syncthreads();
  for (int st = 128; st > 0; st >>= 1) { if (tid < st) red[tid] = fmaxf(red[tid], red[tid + st]); __syncthreads(); }
  if (tid == 0) { gsh = red[0] + EPSF; g2[b] = gsh; }
  __syncthreads();
  const float qs = 127.f / gsh;
  for (int i = tid; i < CD; i += 256) {
    float v = (i < QD) ? query[b * QD + i] : ctxv[b * QD + (i - QD)];
    aqc[b * CD + i] = (signed char)fminf(fmaxf(rintf(v * qs), -128.f), 127.f);
    aqc[(b + 8) * CD + i] = 0;
  }
}

extern "C" void kernel_launch(void* const* d_in, const int* in_sizes, int n_in,
                              void* d_out, int out_size, void* d_ws, size_t ws_size,
                              hipStream_t stream) {
  (void)in_sizes; (void)n_in; (void)out_size; (void)ws_size;
  const float* query = (const float*)d_in[0];
  const float* cmem  = (const float*)d_in[1];
  const float* w_q   = (const float*)d_in[2];
  const float* b_q   = (const float*)d_in[3];
  const float* w_k   = (const float*)d_in[4];
  const float* b_k   = (const float*)d_in[5];
  const float* w_v   = (const float*)d_in[6];
  const float* b_v   = (const float*)d_in[7];
  const float* w_c   = (const float*)d_in[8];
  const float* b_c   = (const float*)d_in[9];

  char* ws = (char*)d_ws;
  float*       part   = (float*)(ws + WS_PART);
  float*       scales = (float*)(ws + WS_SCALES);
  float*       gq     = (float*)(ws + WS_GQ);
  float*       g2     = (float*)(ws + WS_G2);
  float*       offs   = (float*)(ws + WS_OFFS);
  signed char* aqq    = (signed char*)(ws + WS_AQQ);
  signed char* aqc    = (signed char*)(ws + WS_AQC);
  signed char* wqq    = (signed char*)(ws + WS_WQQ);
  signed char* wqk    = (signed char*)(ws + WS_WQK);
  signed char* wqvT   = (signed char*)(ws + WS_WQVT);
  signed char* wqc    = (signed char*)(ws + WS_WQC);
  float*       qproj  = (float*)(ws + WS_QPROJ);
  float*       pvec   = (float*)(ws + WS_P);
  float*       unorm  = (float*)(ws + WS_UNORM);
  float*       ctxv   = (float*)(ws + WS_CTX);
  float*       partm  = (float*)(ws + WS_PARTM);
  float*       partl  = (float*)(ws + WS_PARTL);
  float*       partu  = (float*)(ws + WS_PARTU);
  float*       out    = (float*)d_out;

  k_abs_partial<<<dim3(256, 4), 256, 0, stream>>>(w_q, w_k, w_v, w_c, part);
  k_scales<<<4, 256, 0, stream>>>(part, scales);
  k_quant_w<<<dim3(1024, 4), 256, 0, stream>>>(w_q, w_k, w_v, w_c, scales, wqq, wqk, wqvT, wqc);

  k_quant_query<<<8, 256, 0, stream>>>(query, aqq, gq);
  k_wmma_proj<<<64, 32, 0, stream>>>(aqq, wqq, gq, scales, 0, b_q, qproj, QD);

  k_offs<<<1, 256, 0, stream>>>(qproj, b_k, offs);
  k_pvec<<<4, 256, 0, stream>>>(qproj, wqk, scales, pvec);

  k_scan<<<dim3(NCHUNK, BB), 256, 0, stream>>>(cmem, pvec, offs, partm, partl, partu);
  k_combine<<<8, 256, 0, stream>>>(partm, partl, partu, unorm);

  k_ctx<<<4, 256, 0, stream>>>(unorm, wqvT, scales, b_v, ctxv);

  k_quant_combined<<<8, 256, 0, stream>>>(query, ctxv, aqc, g2);
  k_wmma_proj<<<64, 32, 0, stream>>>(aqc, wqc, g2, scales, 3, b_c, out, CD);
}
